// LinearFP8Wrapper_88450556494665
// MI455X (gfx1250) — matrix operand order repaired and checked by reading, not compiled, not run on hardware
//
#include <hip/hip_runtime.h>

// ---------------------------------------------------------------------------
// y[m,n] = sum_k x[m,k] * dequant_e4m3(w[n,k]) + bias[n]
// M=8192, K=3072, N=12288 -> compute-bound bf16 GEMM on v_wmma_f32_16x16x32_bf16.
// 128x128 block tile, 8 wave32; per wave 4x2 WMMA tiles (64x32).
// Double-buffered LDS (one barrier / K-step); A tile staged via CDNA5
// global_load_async_to_lds_b128 (ASYNCcnt), B tile dequantized fp8->bf16 in
// registers (v_cvt_pk_f32_fp8) once per 128 rows of reuse.
// ---------------------------------------------------------------------------

typedef __attribute__((ext_vector_type(16))) __bf16 v16bf;
typedef __attribute__((ext_vector_type(8)))  __bf16 v8bf;
typedef __attribute__((ext_vector_type(8)))  float  v8f;
typedef __attribute__((ext_vector_type(4)))  unsigned int u32x4;
typedef __attribute__((ext_vector_type(4)))  int i32x4;

#define M_DIM 8192
#define K_DIM 3072
#define N_DIM 12288
#define KT 32
#define LDT 40                 // padded LDS row stride (bf16): 80B = 20 dwords
#define KTILES (K_DIM / KT)    // 96

union ABFrag { v16bf v; v8bf h[2]; };

// ---- CDNA5 async global->LDS copy (16B per lane), sync fallback ----------
__device__ __forceinline__ void async_copy16(const __bf16* g, __bf16* l) {
#if __has_builtin(__builtin_amdgcn_global_load_async_to_lds_b128)
    __builtin_amdgcn_global_load_async_to_lds_b128(
        (i32x4*)g, (i32x4*)l, 0, 0);
#else
    *(v8bf*)l = *(const v8bf*)g;
#endif
}
__device__ __forceinline__ void wait_async() {
#if __has_builtin(__builtin_amdgcn_s_wait_asynccnt)
    __builtin_amdgcn_s_wait_asynccnt(0);
#endif
}

// ---- fp8 e4m3fn -> f32 (sw fallback only) --------------------------------
__device__ __forceinline__ float fp8_e4m3_to_f32_sw(unsigned int v) {
    unsigned s = (v & 0x80u) << 24;
    unsigned e = (v >> 3) & 0xFu;
    unsigned m = v & 0x7u;
    unsigned bits;
    if (e != 0u) {
        if (e == 15u && m == 7u) bits = s | 0x7FC00000u;
        else                     bits = s | ((e + 120u) << 23) | (m << 20);
    } else if (m != 0u) {
        int p = 31 - __builtin_clz(m);
        bits = s | ((unsigned)(118 + p) << 23) | ((m << (23 - p)) & 0x7FFFFFu);
    } else {
        bits = s;
    }
    union { unsigned u; float f; } cvt; cvt.u = bits;
    return cvt.f;
}

// Convert 16 packed fp8 bytes to 16 bf16 and store to LDS (2x ds_store_b128).
__device__ __forceinline__ void cvt16_store(__bf16* dst, u32x4 w) {
    __bf16 tmp[16];
#pragma unroll
    for (int i = 0; i < 4; ++i) {
#if __has_builtin(__builtin_amdgcn_cvt_pk_f32_fp8)
        auto f0 = __builtin_amdgcn_cvt_pk_f32_fp8(w[i], false);
        auto f1 = __builtin_amdgcn_cvt_pk_f32_fp8(w[i], true);
        tmp[4 * i + 0] = (__bf16)f0[0];
        tmp[4 * i + 1] = (__bf16)f0[1];
        tmp[4 * i + 2] = (__bf16)f1[0];
        tmp[4 * i + 3] = (__bf16)f1[1];
#else
        tmp[4 * i + 0] = (__bf16)fp8_e4m3_to_f32_sw((w[i] >> 0)  & 0xFFu);
        tmp[4 * i + 1] = (__bf16)fp8_e4m3_to_f32_sw((w[i] >> 8)  & 0xFFu);
        tmp[4 * i + 2] = (__bf16)fp8_e4m3_to_f32_sw((w[i] >> 16) & 0xFFu);
        tmp[4 * i + 3] = (__bf16)fp8_e4m3_to_f32_sw((w[i] >> 24) & 0xFFu);
#endif
    }
    v8bf lo, hi;
#pragma unroll
    for (int i = 0; i < 8; ++i) { lo[i] = tmp[i]; hi[i] = tmp[8 + i]; }
    *(v8bf*)(dst)     = lo;
    *(v8bf*)(dst + 8) = hi;
}

__global__ __launch_bounds__(256) void
linear_fp8_bf16_wmma(const __bf16* __restrict__ X,
                     const unsigned char* __restrict__ W,
                     const __bf16* __restrict__ Bias,
                     __bf16* __restrict__ Y) {
    __shared__ __bf16 As[2][128 * LDT];   // 2 x 10 KB
    __shared__ __bf16 Bs[2][128 * LDT];   // 2 x 10 KB

    const int tid  = threadIdx.x;
    const int lane = tid & 31;
    const int l16  = lane & 15;
    const int lh   = lane >> 4;
    const int wave = tid >> 5;

    const int m0 = blockIdx.y * 128;
    const int n0 = blockIdx.x * 128;
    const int waveM = (wave >> 2) * 64;
    const int waveN = (wave & 3) * 32;

    // staging coordinates (one 16B chunk granularity)
    const int aRow = tid >> 2;            // 0..63 (+64 twin)
    const int aCol = (tid & 3) * 8;       // bf16 column chunk
    const int bRow = tid >> 1;            // 0..127
    const int bCol = (tid & 1) * 16;      // fp8 byte column chunk

    const __bf16* Aptr0 = X + (size_t)(m0 + aRow) * K_DIM + aCol;
    const __bf16* Aptr1 = Aptr0 + (size_t)64 * K_DIM;
    const unsigned char* Wptr = W + (size_t)(n0 + bRow) * K_DIM + bCol;

    // ---- prologue: stage tile 0 into buffer 0 ----
    async_copy16(Aptr0, &As[0][aRow * LDT + aCol]);
    async_copy16(Aptr1, &As[0][(aRow + 64) * LDT + aCol]);
    {
        u32x4 w0 = *(const u32x4*)(Wptr);
        cvt16_store(&Bs[0][bRow * LDT + bCol], w0);
    }
    wait_async();
    __syncthreads();

    v8f acc[4][2];
#pragma unroll
    for (int mf = 0; mf < 4; ++mf)
#pragma unroll
        for (int nf = 0; nf < 2; ++nf)
#pragma unroll
            for (int r = 0; r < 8; ++r) acc[mf][nf][r] = 0.0f;

    int buf = 0;
    for (int kt = 0; kt < KTILES; ++kt) {
        const bool more = (kt + 1) < KTILES;
        u32x4 nw;
        if (more) {
            const int kk = (kt + 1) * KT;
            // async A copy straight into the other LDS buffer (no VGPR staging)
            async_copy16(Aptr0 + kk, &As[buf ^ 1][aRow * LDT + aCol]);
            async_copy16(Aptr1 + kk, &As[buf ^ 1][(aRow + 64) * LDT + aCol]);
            nw = *(const u32x4*)(Wptr + kk);       // fp8 weights -> regs
        }

        // LDS -> fragments.
        ABFrag afr[4], bfr[2];
#pragma unroll
        for (int mf = 0; mf < 4; ++mf) {
            const __bf16* p = &As[buf][(waveM + mf * 16 + l16) * LDT + lh * 8];
            afr[mf].h[0] = *(const v8bf*)(p);
            afr[mf].h[1] = *(const v8bf*)(p + 16);
        }
#pragma unroll
        for (int nf = 0; nf < 2; ++nf) {
            const __bf16* p = &Bs[buf][(waveN + nf * 16 + l16) * LDT + lh * 8];
            bfr[nf].h[0] = *(const v8bf*)(p);
            bfr[nf].h[1] = *(const v8bf*)(p + 16);
        }

#pragma unroll
        for (int mf = 0; mf < 4; ++mf)
#pragma unroll
            for (int nf = 0; nf < 2; ++nf)
                acc[mf][nf] = __builtin_amdgcn_wmma_f32_16x16x32_bf16(
                    false, afr[mf].v, false, bfr[nf].v,
                    (short)0, acc[mf][nf], false, false);

        if (more)
            cvt16_store(&Bs[buf ^ 1][bRow * LDT + bCol], nw);

        wait_async();            // async LDS writes visible before barrier
        __syncthreads();         // single barrier per K-step (double buffered)
        buf ^= 1;
    }

    // ---- epilogue: C layout VGPR r -> M=r / M=8+r by lane half, N=l16 ----
    float bv[2];
#pragma unroll
    for (int nf = 0; nf < 2; ++nf)
        bv[nf] = (float)Bias[n0 + waveN + nf * 16 + l16];

    const int rowBase = m0 + waveM + (lh << 3);
    const int colBase = n0 + waveN + l16;
#pragma unroll
    for (int mf = 0; mf < 4; ++mf)
#pragma unroll
        for (int nf = 0; nf < 2; ++nf)
#pragma unroll
            for (int r = 0; r < 8; ++r) {
                const int row = rowBase + mf * 16 + r;
                Y[(size_t)row * N_DIM + colBase + nf * 16] =
                    (__bf16)(acc[mf][nf][r] + bv[nf]);
            }
}

extern "C" void kernel_launch(void* const* d_in, const int* in_sizes, int n_in,
                              void* d_out, int out_size, void* d_ws, size_t ws_size,
                              hipStream_t stream) {
    const __bf16*        X = (const __bf16*)d_in[0];
    const unsigned char* W = (const unsigned char*)d_in[1];
    const __bf16*        B = (const __bf16*)d_in[2];
    __bf16*              Y = (__bf16*)d_out;

    dim3 grid(N_DIM / 128, M_DIM / 128, 1);   // 96 x 64 workgroups
    linear_fp8_bf16_wmma<<<grid, dim3(256), 0, stream>>>(X, W, B, Y);
}
